// GraphGPS_74010876444910
// MI455X (gfx1250) — compile-verified, matrix operand order checked by reading, not run on hardware
//
#include <hip/hip_runtime.h>

// Problem constants (match reference)
#define BNODES 8192
#define HDIM   256
#define LAYERS 4
#define NHEADS 8
#define DHEAD  32
#define NEDGE  131072
#define EPSBN  1e-5f
#define LSTR   40      // LDS row stride (bf16 elems): 80B, 16B aligned, conflict-free
#define AVSTR  136     // V-transposed LDS stride: 272B, 16B aligned

typedef unsigned int  uint;
typedef unsigned short ushort;
typedef __attribute__((ext_vector_type(16))) __bf16 v16bf;
typedef __attribute__((ext_vector_type(8)))  float  v8f;

union Frag { uint4 u[2]; v16bf v; };

__device__ inline ushort f2bf(float f) {
    uint u = __float_as_uint(f);
    return (ushort)((u + 0x7fffu + ((u >> 16) & 1u)) >> 16);
}
__device__ inline float bf2f(ushort h) { return __uint_as_float(((uint)h) << 16); }
__device__ inline float gelu_(float x) { return 0.5f * x * (1.0f + erff(x * 0.70710678118654752f)); }

// ---------------- elementwise utility kernels ----------------
__global__ void k_zero(float* p, int n) {
    int i = blockIdx.x * 256 + threadIdx.x;
    if (i < n) p[i] = 0.f;
}
__global__ void k_cvt(const float* __restrict__ a, ushort* __restrict__ b, int n) {
    int i = blockIdx.x * 256 + threadIdx.x;
    if (i < n) b[i] = f2bf(a[i]);
}
// zero-pad columns: src fp32 [rows, sc] -> dst bf16 [rows, 32]
__global__ void k_pad32(const float* __restrict__ src, ushort* __restrict__ dst,
                        int rows, int sc) {
    int i = blockIdx.x * 256 + threadIdx.x;
    if (i >= rows * 32) return;
    int row = i >> 5, col = i & 31;
    dst[i] = (col < sc) ? f2bf(src[row * sc + col]) : (ushort)0;
}
__global__ void k_addb(const float* a, const float* b, float* c, int n) {
    int i = blockIdx.x * 256 + threadIdx.x;
    if (i < n) c[i] = a[i] + b[i];
}
__global__ void k_gelu_ip(float* p, int n) {
    int i = blockIdx.x * 256 + threadIdx.x;
    if (i < n) p[i] = gelu_(p[i]);
}

// ---------------- bf16 WMMA GEMM: C = act(A[8192,K] @ W[M,K]^T + bias) ----------------
// FLAGS: 1=GELU, 2=accumulate existing Cf, 4=write fp32 Cf, 8=write bf16 Cb, 16=has bias
// Requires: K % 32 == 0, M % 64 == 0 (guaranteed by caller).
template <int FLAGS>
__global__ __launch_bounds__(256, 2) void k_gemm(
    const ushort* __restrict__ A, const ushort* __restrict__ W,
    const float* __restrict__ bias, float* __restrict__ Cf, ushort* __restrict__ Cb,
    int K, int M)
{
    __shared__ ushort sA[128 * LSTR];
    __shared__ ushort sB[64 * LSTR];
    const int tid  = threadIdx.x;
    const int lane = tid & 31;
    const int wave = tid >> 5;
    const int wr   = wave >> 1;     // 0..3
    const int wc   = wave & 1;      // 0..1
    const int r0   = blockIdx.y * 128;
    const int c0   = blockIdx.x * 64;

    v8f acc[2][2];
#pragma unroll
    for (int i = 0; i < 2; i++)
#pragma unroll
        for (int j = 0; j < 2; j++)
#pragma unroll
            for (int g = 0; g < 8; g++) acc[i][j][g] = 0.f;

    const int rmap = lane & 15;
    const int koff = (lane < 16) ? 0 : 8;

    // per-thread staging addresses
    const int arow = tid >> 1, ahalf = tid & 1;
    const int brow = tid >> 2, bq = tid & 3;
    const ushort* aptr = A + (long)(r0 + arow) * K + ahalf * 16;
    const ushort* bptr = W + (long)(c0 + brow) * K + bq * 8;
    ushort* asl = sA + arow * LSTR + ahalf * 16;
    ushort* bsl = sB + brow * LSTR + bq * 8;

    for (int k0 = 0; k0 < K; k0 += 32) {
        uint4 a0 = *(const uint4*)(aptr);
        uint4 a1 = *(const uint4*)(aptr + 8);
        uint4 b0 = *(const uint4*)(bptr);
        __builtin_prefetch(aptr + 32, 0, 0);   // next k-step tile -> global_prefetch_b8
        __builtin_prefetch(bptr + 32, 0, 0);
        aptr += 32; bptr += 32;
        *(uint4*)(asl)     = a0;
        *(uint4*)(asl + 8) = a1;
        *(uint4*)(bsl)     = b0;
        __syncthreads();

        Frag fa[2], fb[2];
#pragma unroll
        for (int i = 0; i < 2; i++) {
            const ushort* p = sA + (wr * 32 + i * 16 + rmap) * LSTR + koff;
            fa[i].u[0] = *(const uint4*)(p);
            fa[i].u[1] = *(const uint4*)(p + 16);
        }
#pragma unroll
        for (int j = 0; j < 2; j++) {
            const ushort* p = sB + (wc * 32 + j * 16 + rmap) * LSTR + koff;
            fb[j].u[0] = *(const uint4*)(p);
            fb[j].u[1] = *(const uint4*)(p + 16);
        }
#pragma unroll
        for (int i = 0; i < 2; i++)
#pragma unroll
            for (int j = 0; j < 2; j++)
                acc[i][j] = __builtin_amdgcn_wmma_f32_16x16x32_bf16(
                    false, fa[i].v, false, fb[j].v, (short)0, acc[i][j], false, false);
        __syncthreads();
    }

    const int rhalf = lane >> 4, cidx = lane & 15;
#pragma unroll
    for (int i = 0; i < 2; i++)
#pragma unroll
        for (int j = 0; j < 2; j++) {
            int col = c0 + wc * 32 + j * 16 + cidx;
            float bv = (FLAGS & 16) ? bias[col] : 0.f;
#pragma unroll
            for (int g = 0; g < 8; g++) {
                int row = r0 + wr * 32 + i * 16 + g + 8 * rhalf;
                long idx = (long)row * M + col;
                float v = acc[i][j][g];
                if (FLAGS & 2) v += Cf[idx];
                v += bv;
                if (FLAGS & 1) v = gelu_(v);
                if (FLAGS & 4) Cf[idx] = v;
                if (FLAGS & 8) Cb[idx] = f2bf(v);
            }
        }
}

// ---------------- flash attention (per (graph, head, 128-row q block)) ----------------
__global__ __launch_bounds__(256, 2) void k_attn(const ushort* __restrict__ qkv,
                                                 ushort* __restrict__ o)
{
    __shared__ ushort sK[128 * LSTR];          // current 128 keys x DH
    __shared__ ushort sVt[32 * AVSTR];         // V transposed: DH x 128 keys
    __shared__ ushort sP[8][16 * LSTR];        // per-wave P staging (C-layout -> A-layout)
    const int tid = threadIdx.x, lane = tid & 31, wave = tid >> 5;
    const int bid = blockIdx.x;                // b*64 + nh*8 + qb
    const int qb = bid & 7, nh = (bid >> 3) & 7, b = bid >> 6;
    const long rowbase = (long)b * 1024;
    const int rmap = lane & 15;
    const int koff = (lane < 16) ? 0 : 8;

    // Q fragment (constant over key loop)
    const long qrow = rowbase + qb * 128 + wave * 16 + rmap;
    Frag fq;
    {
        const ushort* p = qkv + qrow * 768 + nh * 32 + koff;
        fq.u[0] = *(const uint4*)(p);
        fq.u[1] = *(const uint4*)(p + 16);
    }
    ushort* sp = sP[wave];
    v8f o0, o1;
    float m[8], l[8];
#pragma unroll
    for (int g = 0; g < 8; g++) { o0[g] = 0.f; o1[g] = 0.f; m[g] = -1e30f; l[g] = 0.f; }
    const float scale = 0.17677669529663689f;   // 1/sqrt(32)
    const int prow = (lane >> 4) * 8;

    // transpose-staging coordinates (fixed per thread): 16 steps of 256 elems
    const int vtd0 = tid >> 7;                  // first dim index contribution
    const int vtr  = tid & 127;                 // key row within block

    for (int kb = 0; kb < 8; kb++) {            // 8 blocks of 128 keys
        __syncthreads();
        { // stage K block (row-major, 128 x 32)
            int row = tid >> 1, half = tid & 1;
            const ushort* src = qkv + (rowbase + kb * 128 + row) * 768 + 256 + nh * 32 + half * 16;
            uint4 a0 = *(const uint4*)(src);
            uint4 a1 = *(const uint4*)(src + 8);
            *(uint4*)(sK + row * LSTR + half * 16)     = a0;
            *(uint4*)(sK + row * LSTR + half * 16 + 8) = a1;
        }
        { // stage V block transposed (DH x 128), exactly 16 strided steps, no bounds test
            const ushort* vsrc = qkv + (rowbase + kb * 128 + vtr) * 768 + 512 + nh * 32;
#pragma unroll
            for (int it = 0; it < 16; it++) {
                int d = vtd0 + it * 2;
                sVt[d * AVSTR + vtr] = vsrc[d];
            }
        }
        __syncthreads();

#pragma unroll
        for (int kt = 0; kt < 4; kt++) {        // 32 keys per step
            Frag fk0, fk1;
            {
                const ushort* p = sK + (kt * 32 + rmap) * LSTR + koff;
                fk0.u[0] = *(const uint4*)(p); fk0.u[1] = *(const uint4*)(p + 16);
            }
            {
                const ushort* p = sK + (kt * 32 + 16 + rmap) * LSTR + koff;
                fk1.u[0] = *(const uint4*)(p); fk1.u[1] = *(const uint4*)(p + 16);
            }
            v8f z;
#pragma unroll
            for (int g = 0; g < 8; g++) z[g] = 0.f;
            v8f s0 = __builtin_amdgcn_wmma_f32_16x16x32_bf16(false, fq.v, false, fk0.v, (short)0, z, false, false);
            v8f s1 = __builtin_amdgcn_wmma_f32_16x16x32_bf16(false, fq.v, false, fk1.v, (short)0, z, false, false);

            // online softmax per row (rows live across 16-lane groups)
#pragma unroll
            for (int g = 0; g < 8; g++) {
                float a0 = s0[g] * scale, a1 = s1[g] * scale;
                float mx = fmaxf(a0, a1);
#pragma unroll
                for (int d = 1; d < 16; d <<= 1) mx = fmaxf(mx, __shfl_xor(mx, d, 16));
                float mn = fmaxf(m[g], mx);
                float alpha = __expf(m[g] - mn);
                float p0 = __expf(a0 - mn), p1 = __expf(a1 - mn);
                float ps = p0 + p1;
#pragma unroll
                for (int d = 1; d < 16; d <<= 1) ps += __shfl_xor(ps, d, 16);
                l[g] = l[g] * alpha + ps;
                m[g] = mn;
                o0[g] *= alpha; o1[g] *= alpha;
                sp[(g + prow) * LSTR + (lane & 15)]      = f2bf(p0);
                sp[(g + prow) * LSTR + 16 + (lane & 15)] = f2bf(p1);
            }
            // reload P in A-layout (same-wave LDS ops are in order)
            Frag fp;
            {
                const ushort* p = sp + rmap * LSTR + koff;
                fp.u[0] = *(const uint4*)(p); fp.u[1] = *(const uint4*)(p + 16);
            }
            Frag fv0, fv1;   // B fragments: column n = head dim d, rows = 32 keys
            {
                const ushort* p = sVt + rmap * AVSTR + kt * 32 + koff;
                fv0.u[0] = *(const uint4*)(p); fv0.u[1] = *(const uint4*)(p + 16);
            }
            {
                const ushort* p = sVt + (16 + rmap) * AVSTR + kt * 32 + koff;
                fv1.u[0] = *(const uint4*)(p); fv1.u[1] = *(const uint4*)(p + 16);
            }
            o0 = __builtin_amdgcn_wmma_f32_16x16x32_bf16(false, fp.v, false, fv0.v, (short)0, o0, false, false);
            o1 = __builtin_amdgcn_wmma_f32_16x16x32_bf16(false, fp.v, false, fv1.v, (short)0, o1, false, false);
        }
    }
#pragma unroll
    for (int g = 0; g < 8; g++) {
        long row = rowbase + qb * 128 + wave * 16 + g + 8 * (lane >> 4);
        float inv = 1.f / l[g];
        o[row * 256 + nh * 32 + (lane & 15)]      = f2bf(o0[g] * inv);
        o[row * 256 + nh * 32 + 16 + (lane & 15)] = f2bf(o1[g] * inv);
    }
}

// ---------------- sparse GCN message scatter ----------------
__global__ void k_edge(const int* __restrict__ er, const int* __restrict__ ec,
                       const float* __restrict__ ev, const float* __restrict__ tmp,
                       float* __restrict__ acc, int E)
{
    long gid = (long)blockIdx.x * 256 + threadIdx.x;
    int e = (int)(gid >> 6);
    if (e >= E) return;
    int c = (int)(gid & 63) << 2;
    float v = ev[e];
    const float4 t = *(const float4*)(tmp + (long)ec[e] * HDIM + c);
    float* dst = acc + (long)er[e] * HDIM + c;
    atomicAdd(dst + 0, v * t.x);
    atomicAdd(dst + 1, v * t.y);
    atomicAdd(dst + 2, v * t.z);
    atomicAdd(dst + 3, v * t.w);
}

// ---------------- BatchNorm (training mode, biased variance over 8192 nodes) ----------------
__global__ void k_bn_stats(const float* __restrict__ h, const float* __restrict__ d,
                           float* __restrict__ sums)
{
    int c = threadIdx.x;                 // 256 cols
    int r0 = blockIdx.x * 32;            // 256 blocks x 32 rows
    float s = 0.f, s2 = 0.f;
#pragma unroll 8
    for (int i = 0; i < 32; i++) {
        long idx = (long)(r0 + i) * HDIM + c;
        float x = h[idx] + d[idx];
        s += x; s2 += x * x;
    }
    atomicAdd(&sums[c], s);
    atomicAdd(&sums[HDIM + c], s2);
}
__global__ void k_bn_norm(float* __restrict__ h, const float* __restrict__ d,
                          const float* __restrict__ sums, const float* __restrict__ g,
                          const float* __restrict__ b, ushort* __restrict__ hbf)
{
    long idx = (long)blockIdx.x * 256 + threadIdx.x;
    int c = (int)(idx & (HDIM - 1));
    float mean = sums[c] * (1.f / BNODES);
    float var  = sums[HDIM + c] * (1.f / BNODES) - mean * mean;
    float x = h[idx] + d[idx];
    float y = (x - mean) * rsqrtf(var + EPSBN) * g[c] + b[c];
    h[idx] = y;
    hbf[idx] = f2bf(y);
}

// ---------------- head layer 2: dot + sigmoid ----------------
__global__ void k_head2(const ushort* __restrict__ mid, const float* __restrict__ w,
                        const float* __restrict__ b, float* __restrict__ out, int rows)
{
    int r = blockIdx.x * 256 + threadIdx.x;
    if (r >= rows) return;
    float s = b[0];
    const ushort* p = mid + (long)r * 128;
#pragma unroll 8
    for (int k = 0; k < 128; k++) s += bf2f(p[k]) * w[k];
    out[r] = 1.f / (1.f + __expf(-s));
}

// ---------------- host-side orchestration ----------------
extern "C" void kernel_launch(void* const* d_in, const int* in_sizes, int n_in,
                              void* d_out, int out_size, void* d_ws, size_t ws_size,
                              hipStream_t stream)
{
    (void)in_sizes; (void)n_in; (void)out_size; (void)ws_size;
    const float* x         = (const float*)d_in[0];
    const int*   erow      = (const int*)d_in[1];
    const int*   ecol      = (const int*)d_in[2];
    const float* eval      = (const float*)d_in[3];
    const float* pe        = (const float*)d_in[4];
    const float* W_in      = (const float*)d_in[5];
    const float* b_in      = (const float*)d_in[6];
    const float* W_pe      = (const float*)d_in[7];
    const float* b_pe      = (const float*)d_in[8];
    const float* gcn_W     = (const float*)d_in[9];
    const float* attn_in_W = (const float*)d_in[10];
    const float* attn_in_b = (const float*)d_in[11];
    const float* attn_out_W= (const float*)d_in[12];
    const float* attn_out_b= (const float*)d_in[13];
    const float* bn_g      = (const float*)d_in[14];
    const float* bn_b      = (const float*)d_in[15];
    const float* ffn_W1    = (const float*)d_in[16];
    const float* ffn_b1    = (const float*)d_in[17];
    const float* ffn_W2    = (const float*)d_in[18];
    const float* ffn_b2    = (const float*)d_in[19];
    const float* head_W1   = (const float*)d_in[20];
    const float* head_b1   = (const float*)d_in[21];
    const float* head_W2   = (const float*)d_in[22];
    const float* head_b2   = (const float*)d_in[23];

    char* w = (char*)d_ws;
    auto alloc = [&](size_t bytes) -> char* {
        char* p = w; w += (bytes + 255) & ~(size_t)255; return p;
    };
    float*  h     = (float*)alloc((size_t)BNODES * HDIM * 4);
    ushort* hbf   = (ushort*)alloc((size_t)BNODES * HDIM * 2);
    float*  delta = (float*)alloc((size_t)BNODES * HDIM * 4);
    float*  tmp   = (float*)alloc((size_t)BNODES * HDIM * 4);
    ushort* qkvb  = (ushort*)alloc((size_t)BNODES * 768 * 2);
    ushort* attnb = (ushort*)alloc((size_t)BNODES * HDIM * 2);
    ushort* ffnb  = (ushort*)alloc((size_t)BNODES * 1024 * 2);
    ushort* midb  = (ushort*)alloc((size_t)BNODES * 128 * 2);
    ushort* xbf   = (ushort*)alloc((size_t)BNODES * 128 * 2);
    ushort* pepad = (ushort*)alloc((size_t)BNODES * 32 * 2);
    float*  biasc = (float*)alloc(HDIM * 4);
    float*  bns   = (float*)alloc(2 * HDIM * 4);
    ushort* Winb  = (ushort*)alloc(256 * 128 * 2);
    ushort* Wpep  = (ushort*)alloc(256 * 32 * 2);
    ushort* gcnb  = (ushort*)alloc((size_t)4 * 256 * 256 * 2);
    ushort* ainb  = (ushort*)alloc((size_t)4 * 768 * 256 * 2);
    ushort* aoutb = (ushort*)alloc((size_t)4 * 256 * 256 * 2);
    ushort* f1b   = (ushort*)alloc((size_t)4 * 1024 * 256 * 2);
    ushort* f2b   = (ushort*)alloc((size_t)4 * 256 * 1024 * 2);
    ushort* h1b   = (ushort*)alloc(128 * 256 * 2);

    auto cvt = [&](const float* s, ushort* dst, int n) {
        k_cvt<<<(n + 255) / 256, 256, 0, stream>>>(s, dst, n);
    };
    cvt(x, xbf, BNODES * 128);
    cvt(W_in, Winb, 256 * 128);
    cvt(gcn_W, gcnb, 4 * 256 * 256);
    cvt(attn_in_W, ainb, 4 * 768 * 256);
    cvt(attn_out_W, aoutb, 4 * 256 * 256);
    cvt(ffn_W1, f1b, 4 * 1024 * 256);
    cvt(ffn_W2, f2b, 4 * 256 * 1024);
    cvt(head_W1, h1b, 128 * 256);
    // zero-pad PE operands to K=32 so all GEMMs take the branch-free staging path
    k_pad32<<<(BNODES * 32 + 255) / 256, 256, 0, stream>>>(pe, pepad, BNODES, 8);
    k_pad32<<<(256 * 32 + 255) / 256, 256, 0, stream>>>(W_pe, Wpep, 256, 8);
    k_addb<<<1, 256, 0, stream>>>(b_in, b_pe, biasc, HDIM);

    auto gemm = [&](const ushort* A, const ushort* Wm, const float* bias,
                    float* Cf, ushort* Cb, int K, int M, int flags) {
        dim3 g(M / 64, BNODES / 128);
        switch (flags) {
            case 4:  k_gemm<4><<<g, 256, 0, stream>>>(A, Wm, bias, Cf, Cb, K, M); break;
            case 14: k_gemm<14><<<g, 256, 0, stream>>>(A, Wm, bias, Cf, Cb, K, M); break;
            case 20: k_gemm<20><<<g, 256, 0, stream>>>(A, Wm, bias, Cf, Cb, K, M); break;
            case 24: k_gemm<24><<<g, 256, 0, stream>>>(A, Wm, bias, Cf, Cb, K, M); break;
            case 25: k_gemm<25><<<g, 256, 0, stream>>>(A, Wm, bias, Cf, Cb, K, M); break;
            default: break;
        }
    };

    // input embedding: h = x @ W_in^T + pe @ W_pe^T + (b_in + b_pe)
    gemm(xbf, Winb, biasc, h, nullptr, 128, 256, 4 | 16);
    gemm(pepad, Wpep, nullptr, h, hbf, 32, 256, 2 | 4 | 8);

    for (int l = 0; l < LAYERS; l++) {
        // local GCN
        gemm(hbf, gcnb + (size_t)l * 256 * 256, nullptr, tmp, nullptr, 256, 256, 4);
        k_zero<<<(BNODES * HDIM) / 256, 256, 0, stream>>>(delta, BNODES * HDIM);
        k_edge<<<(NEDGE * 64) / 256, 256, 0, stream>>>(erow, ecol, eval, tmp, delta, NEDGE);
        k_gelu_ip<<<(BNODES * HDIM) / 256, 256, 0, stream>>>(delta, BNODES * HDIM);
        k_zero<<<2, 256, 0, stream>>>(bns, 2 * HDIM);
        k_bn_stats<<<256, 256, 0, stream>>>(h, delta, bns);
        k_bn_norm<<<BNODES, 256, 0, stream>>>(h, delta, bns,
                                              bn_g + (l * 3 + 0) * HDIM, bn_b + (l * 3 + 0) * HDIM, hbf);
        // global attention
        gemm(hbf, ainb + (size_t)l * 768 * 256, attn_in_b + l * 768, nullptr, qkvb, 256, 768, 8 | 16);
        k_attn<<<512, 256, 0, stream>>>(qkvb, attnb);
        gemm(attnb, aoutb + (size_t)l * 256 * 256, attn_out_b + l * 256, delta, nullptr, 256, 256, 4 | 16);
        k_zero<<<2, 256, 0, stream>>>(bns, 2 * HDIM);
        k_bn_stats<<<256, 256, 0, stream>>>(h, delta, bns);
        k_bn_norm<<<BNODES, 256, 0, stream>>>(h, delta, bns,
                                              bn_g + (l * 3 + 1) * HDIM, bn_b + (l * 3 + 1) * HDIM, hbf);
        // FFN
        gemm(hbf, f1b + (size_t)l * 1024 * 256, ffn_b1 + l * 1024, nullptr, ffnb, 256, 1024, 1 | 8 | 16);
        gemm(ffnb, f2b + (size_t)l * 256 * 1024, ffn_b2 + l * 256, delta, nullptr, 1024, 256, 4 | 16);
        k_zero<<<2, 256, 0, stream>>>(bns, 2 * HDIM);
        k_bn_stats<<<256, 256, 0, stream>>>(h, delta, bns);
        k_bn_norm<<<BNODES, 256, 0, stream>>>(h, delta, bns,
                                              bn_g + (l * 3 + 2) * HDIM, bn_b + (l * 3 + 2) * HDIM, hbf);
    }
    // head
    gemm(hbf, h1b, head_b1, nullptr, midb, 256, 128, 1 | 8 | 16);
    k_head2<<<BNODES / 256, 256, 0, stream>>>(midb, head_W2, head_b2, (float*)d_out, BNODES);
}
